// RWKV_Block_63144609185890
// MI455X (gfx1250) — matrix operand order repaired and checked by reading, not compiled, not run on hardware
//
#include <hip/hip_runtime.h>

// RWKV block for MI455X (gfx1250, wave32, WMMA).
// Compute-bound (~0.96 TFLOP of GEMM vs ~0.7 GB of traffic) => all 7 GEMMs go
// through v_wmma_f32_16x16x32_f16 with f32 accumulators; LN/exp/sigmoid stay f32.
// Weight/B tiles (and pure-f16 A tiles) are staged into LDS with
// GLOBAL_LOAD_ASYNC_TO_LDS_B128 (ASYNCcnt) so the DMA overlaps the VALU work
// that builds the token-shift+mu-mixed A tiles.
// Workspace layout (needs ~312 MB):
//   [0,56MB)    : 7 weight matrices converted to f16 (8 MB each)
//   [56,184MB)  : h (f32, 16384x2048)    -- LN output, reused for LN2
//   [184,248MB) : s1 (f16)               -- rwkv, later r2
//   [248,312MB) : s2 (f16)               -- k2
// x1 (time-mixed residual) is staged directly in d_out and updated in place.

constexpr int kD    = 2048;   // model dim
constexpr int kT    = 2048;   // sequence length (power of two)
constexpr int kRows = 16384;  // B*T
constexpr int kBK   = 32;     // WMMA K per step (f16)
constexpr int kLPW  = 20;     // LDS u32 words per tile row: 16 data + 4 pad (80B rows)

typedef _Float16 v16h __attribute__((ext_vector_type(16)));
typedef _Float16 v2h  __attribute__((ext_vector_type(2)));
typedef float    v8f  __attribute__((ext_vector_type(8)));
typedef float    v4f  __attribute__((ext_vector_type(4)));
typedef unsigned v4u  __attribute__((ext_vector_type(4)));
typedef unsigned v2u  __attribute__((ext_vector_type(2)));

struct Frag32 { v4u lo, hi; };

static __device__ __forceinline__ unsigned pkh(float a, float b) {
  v2h h; h[0] = (_Float16)a; h[1] = (_Float16)b;
  return __builtin_bit_cast(unsigned, h);
}

// Generic LDS pointer -> 32-bit LDS byte address (flat->LDS mapping keeps [31:0]).
static __device__ __forceinline__ unsigned lds_addr(const void* p) {
  return (unsigned)(uintptr_t)p;
}

// Async DMA: 16 bytes per lane, global (SGPR base + 32-bit lane offset) -> LDS.
// Tracked by ASYNCcnt; drain with s_wait_asynccnt before the tile barrier.
static __device__ __forceinline__ void async_g2l_b128(unsigned ldsByteAddr,
                                                      unsigned long long gbase,
                                                      unsigned goffBytes) {
  asm volatile("global_load_async_to_lds_b128 %0, %1, %2 offset:0"
               :: "v"(ldsByteAddr), "v"(goffBytes), "s"(gbase)
               : "memory");
}

static __device__ __forceinline__ void wait_async0() {
  asm volatile("s_wait_asynccnt 0x0" ::: "memory");
}

// A fragment (16x32 f16). ISA layout: lane<16 holds M=lane, K pairs {0..3,8..11};
// lane>=16 holds M=lane-16, K pairs {4..7,12..15}. With 80B row pitch this is two
// ds_load_b128 at rowbase + half*16 and +32.
static __device__ __forceinline__ v16h frag_a(const unsigned* tile, int row, int lane) {
  const int m  = row + (lane & 15);
  const int hl = lane >> 4;
  const char* p = (const char*)tile + m * 80 + hl * 16;
  Frag32 f;
  f.lo = *(const v4u*)p;
  f.hi = *(const v4u*)(p + 32);
  return __builtin_bit_cast(v16h, f);
}

// B fragment (32x16 f16). lane<16: N=lane, K {0..7,16..23};
// lane>=16: same N, K {8..15,24..31}. Two ds_load_b128 at rowbase + half*16 and +32.
static __device__ __forceinline__ v16h frag_b(const unsigned* tile, int row, int lane) {
  const int n  = row + (lane & 15);
  const int hl = lane >> 4;
  const char* p = (const char*)tile + n * 80 + hl * 16;
  Frag32 f;
  f.lo = *(const v4u*)p;
  f.hi = *(const v4u*)(p + 32);
  return __builtin_bit_cast(v16h, f);
}

static __device__ __forceinline__ v8f wmma_f16(v16h a, v16h b, v8f c) {
  return __builtin_amdgcn_wmma_f32_16x16x32_f16(false, a, false, b, (short)0, c,
                                                false, false);
}

// Stage a 128x32 f16 weight tile into LDS via async DMA (2 x b128 per thread).
static __device__ __forceinline__ void fill_B_async(unsigned* dst,
                                                    const _Float16* __restrict__ W,
                                                    size_t colBase, int k0, int tid) {
  const unsigned long long gbase = (unsigned long long)(uintptr_t)W;
#pragma unroll
  for (int it = 0; it < 2; ++it) {
    const int idx = tid + it * 256;
    const int r   = idx >> 2;
    const int cq  = (idx & 3) * 8;
    const unsigned goff =
        (unsigned)(((colBase + r) * (size_t)kD + (size_t)(k0 + cq)) * 2u);
    async_g2l_b128(lds_addr(&dst[r * kLPW + (cq >> 1)]), gbase, goff);
  }
}

// Stage a 64x32 f16 activation tile into LDS via async DMA (1 x b128 per thread).
static __device__ __forceinline__ void fill_A16_async(unsigned* dst,
                                                      const _Float16* __restrict__ A,
                                                      size_t rowBase, int k0, int tid) {
  const int r  = tid >> 2;
  const int cq = (tid & 3) * 8;
  const unsigned goff =
      (unsigned)(((rowBase + r) * (size_t)kD + (size_t)(k0 + cq)) * 2u);
  async_g2l_b128(lds_addr(&dst[r * kLPW + (cq >> 1)]),
                 (unsigned long long)(uintptr_t)A, goff);
}

// ---------------------------------------------------------------------------
// LayerNorm over rows of 2048 (one 256-thread block per row, 8 elems/thread).
// ---------------------------------------------------------------------------
__global__ void __launch_bounds__(256) ln_rows(const float* __restrict__ x,
                                               const float* __restrict__ g,
                                               const float* __restrict__ b,
                                               float* __restrict__ out) {
  __shared__ float rs[256];
  __shared__ float rq[256];
  const int tid = threadIdx.x;
  const size_t row = blockIdx.x;
  const float* xr = x + row * (size_t)kD;
  const int base = tid * 8;
  v4f a0 = *(const v4f*)(xr + base);
  v4f a1 = *(const v4f*)(xr + base + 4);
  float s = a0[0]+a0[1]+a0[2]+a0[3] + a1[0]+a1[1]+a1[2]+a1[3];
  float q = a0[0]*a0[0]+a0[1]*a0[1]+a0[2]*a0[2]+a0[3]*a0[3]
          + a1[0]*a1[0]+a1[1]*a1[1]+a1[2]*a1[2]+a1[3]*a1[3];
  rs[tid] = s; rq[tid] = q;
  __syncthreads();
  for (int off = 128; off > 0; off >>= 1) {
    if (tid < off) { rs[tid] += rs[tid + off]; rq[tid] += rq[tid + off]; }
    __syncthreads();
  }
  const float mean = rs[0] * (1.f / kD);
  const float var  = rq[0] * (1.f / kD) - mean * mean;
  const float inv  = rsqrtf(var + 1e-5f);
  v4f g0 = *(const v4f*)(g + base), g1 = *(const v4f*)(g + base + 4);
  v4f b0 = *(const v4f*)(b + base), b1 = *(const v4f*)(b + base + 4);
  v4f o0, o1;
#pragma unroll
  for (int j = 0; j < 4; ++j) {
    o0[j] = (a0[j] - mean) * inv * g0[j] + b0[j];
    o1[j] = (a1[j] - mean) * inv * g1[j] + b1[j];
  }
  float* orow = out + row * (size_t)kD;
  *(v4f*)(orow + base)     = o0;
  *(v4f*)(orow + base + 4) = o1;
}

// ---------------------------------------------------------------------------
// f32 -> f16 weight conversion (8 elems/thread, exact multiples).
// ---------------------------------------------------------------------------
__global__ void __launch_bounds__(256) cvt_f32_f16(const float* __restrict__ s,
                                                   _Float16* __restrict__ d) {
  const size_t i = ((size_t)blockIdx.x * 256 + threadIdx.x) * 8;
  v4f a = *(const v4f*)(s + i);
  v4f b = *(const v4f*)(s + i + 4);
  v4u o;
  o[0] = pkh(a[0], a[1]); o[1] = pkh(a[2], a[3]);
  o[2] = pkh(b[0], b[1]); o[3] = pkh(b[2], b[3]);
  *(v4u*)(d + i) = o;
}

// ---------------------------------------------------------------------------
// Fused r/k/v GEMM: A-tiles built on the fly from h with token-shift + mu mix,
// three weight matrices (async DMA to LDS), epilogue computes
// rwkv = sigmoid(r)*ek*v/(ek+1e-8).
// Block tile 64(M) x 128(N), 8 waves as 4x2, wave tile 16x64 (4 C tiles/output).
// ---------------------------------------------------------------------------
__global__ void __launch_bounds__(256) gemm_rkv(
    const float* __restrict__ h,
    const _Float16* __restrict__ Wr, const _Float16* __restrict__ Wk,
    const _Float16* __restrict__ Wv,
    const float* __restrict__ mur, const float* __restrict__ muk,
    const float* __restrict__ muv, const float* __restrict__ uvec,
    _Float16* __restrict__ rwkv) {
  __shared__ __align__(16) unsigned aA[3][64 * kLPW];
  __shared__ __align__(16) unsigned aB[3][128 * kLPW];
  const int tid = threadIdx.x;
  const int lane = tid & 31, wave = tid >> 5;
  const int waveM = wave >> 1, waveN = wave & 1;
  const size_t rowBase = (size_t)blockIdx.y * 64;
  const size_t colBase = (size_t)blockIdx.x * 128;

  const v8f vz = {0.f,0.f,0.f,0.f,0.f,0.f,0.f,0.f};
  v8f aR[4], aK[4], aV[4];
#pragma unroll
  for (int t = 0; t < 4; ++t) { aR[t] = vz; aK[t] = vz; aV[t] = vz; }

  for (int k0 = 0; k0 < kD; k0 += kBK) {
    __syncthreads();
    // Kick off the weight-tile DMA first so it overlaps the A-mix VALU work.
    fill_B_async(aB[0], Wr, colBase, k0, tid);
    fill_B_async(aB[1], Wk, colBase, k0, tid);
    fill_B_async(aB[2], Wv, colBase, k0, tid);
#pragma unroll
    for (int it = 0; it < 2; ++it) {
      const int idx = tid + it * 256;
      const int r   = idx >> 3;
      const int c4  = (idx & 7) * 4;
      const int kc  = k0 + c4;
      const size_t gm = rowBase + r;
      const int tpos  = (int)(gm & (size_t)(kT - 1));
      const float* hp0 = h + gm * (size_t)kD + kc;
      v4f hv  = *(const v4f*)hp0;
      v4f hpv = {0.f, 0.f, 0.f, 0.f};
      if (tpos != 0) hpv = *(const v4f*)(hp0 - kD);   // token shift, zero at t=0
      if (k0 + kBK < kD) __builtin_prefetch((const void*)(hp0 + kBK), 0, 1);
      const int wo = r * kLPW + (idx & 7) * 2;
      {
        v4f m = *(const v4f*)(mur + kc);
        v2u o; o[0] = pkh(m[0]*hv[0]+(1.f-m[0])*hpv[0], m[1]*hv[1]+(1.f-m[1])*hpv[1]);
               o[1] = pkh(m[2]*hv[2]+(1.f-m[2])*hpv[2], m[3]*hv[3]+(1.f-m[3])*hpv[3]);
        *(v2u*)&aA[0][wo] = o;
      }
      {
        v4f m = *(const v4f*)(muk + kc);
        v2u o; o[0] = pkh(m[0]*hv[0]+(1.f-m[0])*hpv[0], m[1]*hv[1]+(1.f-m[1])*hpv[1]);
               o[1] = pkh(m[2]*hv[2]+(1.f-m[2])*hpv[2], m[3]*hv[3]+(1.f-m[3])*hpv[3]);
        *(v2u*)&aA[1][wo] = o;
      }
      {
        v4f m = *(const v4f*)(muv + kc);
        v2u o; o[0] = pkh(m[0]*hv[0]+(1.f-m[0])*hpv[0], m[1]*hv[1]+(1.f-m[1])*hpv[1]);
               o[1] = pkh(m[2]*hv[2]+(1.f-m[2])*hpv[2], m[3]*hv[3]+(1.f-m[3])*hpv[3]);
        *(v2u*)&aA[2][wo] = o;
      }
    }
    wait_async0();
    __syncthreads();

    const v16h fAr = frag_a(aA[0], waveM * 16, lane);
    const v16h fAk = frag_a(aA[1], waveM * 16, lane);
    const v16h fAv = frag_a(aA[2], waveM * 16, lane);
#pragma unroll
    for (int tn = 0; tn < 4; ++tn) {
      const int nb = waveN * 64 + tn * 16;
      aR[tn] = wmma_f16(fAr, frag_b(aB[0], nb, lane), aR[tn]);
      aK[tn] = wmma_f16(fAk, frag_b(aB[1], nb, lane), aK[tn]);
      aV[tn] = wmma_f16(fAv, frag_b(aB[2], nb, lane), aV[tn]);
    }
  }

  const int hl = lane >> 4, ln = lane & 15;
#pragma unroll
  for (int tn = 0; tn < 4; ++tn) {
    const size_t gn = colBase + waveN * 64 + tn * 16 + ln;
    const float uu = uvec[gn];
#pragma unroll
    for (int v = 0; v < 8; ++v) {
      const size_t gm = rowBase + waveM * 16 + v + 8 * hl;
      const float ek  = __expf(uu + aK[tn][v]);
      const float wkv = (ek * aV[tn][v]) / (ek + 1e-8f);
      const float sg  = 1.f / (1.f + __expf(-aR[tn][v]));
      rwkv[gm * (size_t)kD + gn] = (_Float16)(sg * wkv);
    }
  }
}

// ---------------------------------------------------------------------------
// xout = xin + Af @ W^T   (time-mix output projection + residual)
// ---------------------------------------------------------------------------
__global__ void __launch_bounds__(256) gemm_addx(
    const _Float16* __restrict__ Af, const _Float16* __restrict__ W,
    const float* __restrict__ xin, float* __restrict__ xout) {
  __shared__ __align__(16) unsigned aA[64 * kLPW];
  __shared__ __align__(16) unsigned aB[128 * kLPW];
  const int tid = threadIdx.x;
  const int lane = tid & 31, wave = tid >> 5;
  const int waveM = wave >> 1, waveN = wave & 1;
  const size_t rowBase = (size_t)blockIdx.y * 64;
  const size_t colBase = (size_t)blockIdx.x * 128;

  const v8f vz = {0.f,0.f,0.f,0.f,0.f,0.f,0.f,0.f};
  v8f acc[4];
#pragma unroll
  for (int t = 0; t < 4; ++t) acc[t] = vz;

  for (int k0 = 0; k0 < kD; k0 += kBK) {
    __syncthreads();
    fill_B_async(aB, W, colBase, k0, tid);
    fill_A16_async(aA, Af, rowBase, k0, tid);
    wait_async0();
    __syncthreads();
    const v16h fA = frag_a(aA, waveM * 16, lane);
#pragma unroll
    for (int tn = 0; tn < 4; ++tn)
      acc[tn] = wmma_f16(fA, frag_b(aB, waveN * 64 + tn * 16, lane), acc[tn]);
  }

  const int hl = lane >> 4, ln = lane & 15;
#pragma unroll
  for (int tn = 0; tn < 4; ++tn) {
    const size_t gn = colBase + waveN * 64 + tn * 16 + ln;
#pragma unroll
    for (int v = 0; v < 8; ++v) {
      const size_t gm = rowBase + waveM * 16 + v + 8 * hl;
      const size_t off = gm * (size_t)kD + gn;
      xout[off] = xin[off] + acc[tn][v];
    }
  }
}

// ---------------------------------------------------------------------------
// Fused channel-mix r2/k2 GEMM: A-tiles from h2 with shift+mix; epilogue
// stores r2 = sigmoid(.), k2 = relu(.)^2, both as f16.
// ---------------------------------------------------------------------------
__global__ void __launch_bounds__(256) gemm_rk2(
    const float* __restrict__ h,
    const _Float16* __restrict__ Wr, const _Float16* __restrict__ Wk,
    const float* __restrict__ mur, const float* __restrict__ muk,
    _Float16* __restrict__ r2out, _Float16* __restrict__ k2out) {
  __shared__ __align__(16) unsigned aA[2][64 * kLPW];
  __shared__ __align__(16) unsigned aB[2][128 * kLPW];
  const int tid = threadIdx.x;
  const int lane = tid & 31, wave = tid >> 5;
  const int waveM = wave >> 1, waveN = wave & 1;
  const size_t rowBase = (size_t)blockIdx.y * 64;
  const size_t colBase = (size_t)blockIdx.x * 128;

  const v8f vz = {0.f,0.f,0.f,0.f,0.f,0.f,0.f,0.f};
  v8f aR[4], aK[4];
#pragma unroll
  for (int t = 0; t < 4; ++t) { aR[t] = vz; aK[t] = vz; }

  for (int k0 = 0; k0 < kD; k0 += kBK) {
    __syncthreads();
    fill_B_async(aB[0], Wr, colBase, k0, tid);
    fill_B_async(aB[1], Wk, colBase, k0, tid);
#pragma unroll
    for (int it = 0; it < 2; ++it) {
      const int idx = tid + it * 256;
      const int r   = idx >> 3;
      const int c4  = (idx & 7) * 4;
      const int kc  = k0 + c4;
      const size_t gm = rowBase + r;
      const int tpos  = (int)(gm & (size_t)(kT - 1));
      const float* hp0 = h + gm * (size_t)kD + kc;
      v4f hv  = *(const v4f*)hp0;
      v4f hpv = {0.f, 0.f, 0.f, 0.f};
      if (tpos != 0) hpv = *(const v4f*)(hp0 - kD);
      if (k0 + kBK < kD) __builtin_prefetch((const void*)(hp0 + kBK), 0, 1);
      const int wo = r * kLPW + (idx & 7) * 2;
      {
        v4f m = *(const v4f*)(mur + kc);
        v2u o; o[0] = pkh(m[0]*hv[0]+(1.f-m[0])*hpv[0], m[1]*hv[1]+(1.f-m[1])*hpv[1]);
               o[1] = pkh(m[2]*hv[2]+(1.f-m[2])*hpv[2], m[3]*hv[3]+(1.f-m[3])*hpv[3]);
        *(v2u*)&aA[0][wo] = o;
      }
      {
        v4f m = *(const v4f*)(muk + kc);
        v2u o; o[0] = pkh(m[0]*hv[0]+(1.f-m[0])*hpv[0], m[1]*hv[1]+(1.f-m[1])*hpv[1]);
               o[1] = pkh(m[2]*hv[2]+(1.f-m[2])*hpv[2], m[3]*hv[3]+(1.f-m[3])*hpv[3]);
        *(v2u*)&aA[1][wo] = o;
      }
    }
    wait_async0();
    __syncthreads();

    const v16h fAr = frag_a(aA[0], waveM * 16, lane);
    const v16h fAk = frag_a(aA[1], waveM * 16, lane);
#pragma unroll
    for (int tn = 0; tn < 4; ++tn) {
      const int nb = waveN * 64 + tn * 16;
      aR[tn] = wmma_f16(fAr, frag_b(aB[0], nb, lane), aR[tn]);
      aK[tn] = wmma_f16(fAk, frag_b(aB[1], nb, lane), aK[tn]);
    }
  }

  const int hl = lane >> 4, ln = lane & 15;
#pragma unroll
  for (int tn = 0; tn < 4; ++tn) {
    const size_t gn = colBase + waveN * 64 + tn * 16 + ln;
#pragma unroll
    for (int v = 0; v < 8; ++v) {
      const size_t gm = rowBase + waveM * 16 + v + 8 * hl;
      const size_t off = gm * (size_t)kD + gn;
      r2out[off] = (_Float16)(1.f / (1.f + __expf(-aR[tn][v])));
      const float kk = aK[tn][v];
      const float rl = kk > 0.f ? kk : 0.f;
      k2out[off] = (_Float16)(rl * rl);
    }
  }
}

// ---------------------------------------------------------------------------
// xio = xio + r2 * (k2 @ cm_Wv^T)   (final residual; xio is d_out, in place)
// ---------------------------------------------------------------------------
__global__ void __launch_bounds__(256) gemm_final(
    const _Float16* __restrict__ k2, const _Float16* __restrict__ W,
    const _Float16* __restrict__ r2, float* xio) {
  __shared__ __align__(16) unsigned aA[64 * kLPW];
  __shared__ __align__(16) unsigned aB[128 * kLPW];
  const int tid = threadIdx.x;
  const int lane = tid & 31, wave = tid >> 5;
  const int waveM = wave >> 1, waveN = wave & 1;
  const size_t rowBase = (size_t)blockIdx.y * 64;
  const size_t colBase = (size_t)blockIdx.x * 128;

  const v8f vz = {0.f,0.f,0.f,0.f,0.f,0.f,0.f,0.f};
  v8f acc[4];
#pragma unroll
  for (int t = 0; t < 4; ++t) acc[t] = vz;

  for (int k0 = 0; k0 < kD; k0 += kBK) {
    __syncthreads();
    fill_B_async(aB, W, colBase, k0, tid);
    fill_A16_async(aA, k2, rowBase, k0, tid);
    wait_async0();
    __syncthreads();
    const v16h fA = frag_a(aA, waveM * 16, lane);
#pragma unroll
    for (int tn = 0; tn < 4; ++tn)
      acc[tn] = wmma_f16(fA, frag_b(aB, waveN * 64 + tn * 16, lane), acc[tn]);
  }

  const int hl = lane >> 4, ln = lane & 15;
#pragma unroll
  for (int tn = 0; tn < 4; ++tn) {
    const size_t gn = colBase + waveN * 64 + tn * 16 + ln;
#pragma unroll
    for (int v = 0; v < 8; ++v) {
      const size_t gm = rowBase + waveM * 16 + v + 8 * hl;
      const size_t off = gm * (size_t)kD + gn;
      xio[off] = xio[off] + (float)r2[off] * acc[tn][v];
    }
  }
}

// ---------------------------------------------------------------------------
extern "C" void kernel_launch(void* const* d_in, const int* in_sizes, int n_in,
                              void* d_out, int out_size, void* d_ws, size_t ws_size,
                              hipStream_t stream) {
  (void)in_sizes; (void)n_in; (void)out_size; (void)ws_size;
  const float* x      = (const float*)d_in[0];
  const float* ln1_g  = (const float*)d_in[1];
  const float* ln1_b  = (const float*)d_in[2];
  const float* ln2_g  = (const float*)d_in[3];
  const float* ln2_b  = (const float*)d_in[4];
  // d_in[5] = tm_w: unused (zero state => decay never contributes, per reference)
  const float* tm_u   = (const float*)d_in[6];
  const float* tm_mur = (const float*)d_in[7];
  const float* tm_muk = (const float*)d_in[8];
  const float* tm_muv = (const float*)d_in[9];
  const float* Wsrc[7] = {
      (const float*)d_in[10],  // tm_Wr
      (const float*)d_in[11],  // tm_Wk
      (const float*)d_in[12],  // tm_Wv
      (const float*)d_in[13],  // tm_Wout
      (const float*)d_in[16],  // cm_Wr
      (const float*)d_in[17],  // cm_Wk
      (const float*)d_in[18],  // cm_Wv
  };
  const float* cm_mur = (const float*)d_in[14];
  const float* cm_muk = (const float*)d_in[15];

  char* ws = (char*)d_ws;
  const size_t MB = 1ull << 20;
  _Float16* W16[7];
  for (int i = 0; i < 7; ++i) W16[i] = (_Float16*)(ws + (size_t)i * 8 * MB);
  float*    hbuf = (float*)(ws + 56 * MB);      // 128 MB
  _Float16* s1   = (_Float16*)(ws + 184 * MB);  // 64 MB (rwkv -> r2)
  _Float16* s2   = (_Float16*)(ws + 248 * MB);  // 64 MB (k2)
  float*    xout = (float*)d_out;               // x1 staged here, updated in place

  for (int i = 0; i < 7; ++i)
    cvt_f32_f16<<<2048, 256, 0, stream>>>(Wsrc[i], W16[i]);

  const dim3 gg(16, 256);  // N tiles (2048/128) x M tiles (16384/64)
  ln_rows<<<kRows, 256, 0, stream>>>(x, ln1_g, ln1_b, hbuf);
  gemm_rkv<<<gg, 256, 0, stream>>>(hbuf, W16[0], W16[1], W16[2],
                                   tm_mur, tm_muk, tm_muv, tm_u, s1);
  gemm_addx<<<gg, 256, 0, stream>>>(s1, W16[3], x, xout);
  ln_rows<<<kRows, 256, 0, stream>>>(xout, ln2_g, ln2_b, hbuf);
  gemm_rk2<<<gg, 256, 0, stream>>>(hbuf, W16[4], W16[5], cm_mur, cm_muk, s1, s2);
  gemm_final<<<gg, 256, 0, stream>>>(s2, W16[6], s1, xout);
}
